// MultiHeadedAttention_71648644432026
// MI455X (gfx1250) — matrix operand order repaired and checked by reading, not compiled, not run on hardware
//
#include <hip/hip_runtime.h>

// Problem dims (compile-time)
#define B_  8
#define D_  512
#define N_  1024
#define H_  8
#define HD_ 64

typedef __attribute__((ext_vector_type(16))) __bf16 v16bf;
typedef __attribute__((ext_vector_type(8)))  float  v8f;

union Frag16 {
  v16bf v;
  uint4 q[2];
  unsigned short u[16];
};

__device__ __forceinline__ unsigned short f2bf(float f) {
  unsigned int u = __float_as_uint(f);
  u += 0x7FFFu + ((u >> 16) & 1u);   // round-to-nearest-even
  return (unsigned short)(u >> 16);
}

// ---------------------------------------------------------------------------
// f32 -> bf16 conversion (contiguous), used for W only.
// ---------------------------------------------------------------------------
__global__ void cvt_bf16_kernel(const float* __restrict__ in,
                                unsigned short* __restrict__ out, int n4) {
  int i = blockIdx.x * blockDim.x + threadIdx.x;
  if (i < n4) {
    float4 f = ((const float4*)in)[i];
    ushort4 o;
    o.x = f2bf(f.x); o.y = f2bf(f.y); o.z = f2bf(f.z); o.w = f2bf(f.w);
    ((ushort4*)out)[i] = o;
  }
}

// ---------------------------------------------------------------------------
// f32 [B,D,N] -> bf16 [B,N,D] transpose+convert, 32x32 LDS tiles.
// Makes the GEMM B-fragments contiguous (K innermost per activation row).
// ---------------------------------------------------------------------------
__global__ __launch_bounds__(256) void cvtT_kernel(
    const float* __restrict__ in, unsigned short* __restrict__ out) {
  __shared__ unsigned short t[32][33];       // padded: no bank conflicts
  const int b  = blockIdx.z;
  const int d0 = blockIdx.y * 32;
  const int n0 = blockIdx.x * 32;
  const int tid = threadIdx.x;
  const int r  = tid >> 3;                   // 0..31
  const int c4 = (tid & 7) * 4;              // 0,4,...,28

  float4 f = *(const float4*)&in[((size_t)b * D_ + d0 + r) * N_ + n0 + c4];
  t[r][c4 + 0] = f2bf(f.x);
  t[r][c4 + 1] = f2bf(f.y);
  t[r][c4 + 2] = f2bf(f.z);
  t[r][c4 + 3] = f2bf(f.w);
  __syncthreads();

  ushort4 o;
  o.x = t[c4 + 0][r];
  o.y = t[c4 + 1][r];
  o.z = t[c4 + 2][r];
  o.w = t[c4 + 3][r];
  *(ushort4*)&out[((size_t)b * N_ + n0 + r) * D_ + d0 + c4] = o;
}

// Output layout modes for the projection GEMM
#define OUT_QK  0   // [B,H,N,HD] bf16 (hd innermost) -> score-side fragments
#define OUT_V   1   // [B,H,HD,N] bf16 (m innermost)  -> PV-side fragments
#define OUT_F32 2   // [B,D,N] f32 (final merge into d_out)

#define LDSK 136    // padded LDS row stride (shorts) for the 128-k chunk

// ---------------------------------------------------------------------------
// C[b] = W(DxD) * X[b] + bias ; X given transposed [B,N,D] bf16 ("XT").
// Block: 256 threads (8 waves). Wave w owns rows [m0+16w,+16), cols [n0,+16).
// Staging: [16n][128k] chunk, 1 global b128 -> 1 ds_store_b128 per thread.
// Fragments: A = 2 global b128, B = 2 ds_load_b128 (all contiguous).
// ---------------------------------------------------------------------------
__global__ __launch_bounds__(256) void gemm_bias_kernel(
    const unsigned short* __restrict__ Wbf,   // [D,D] row-major (o,i)
    const unsigned short* __restrict__ XT,    // [B,N,D] bf16
    const float* __restrict__ bias,           // [D]
    unsigned short* __restrict__ out_bf,
    float* __restrict__ out_f32,
    int mode) {
  __shared__ unsigned short ldsX[16 * LDSK]; // [n][k] chunk, padded

  const int tid  = threadIdx.x;
  const int wave = tid >> 5, lane = tid & 31;
  const int col  = lane & 15, hi = lane >> 4;
  const int b  = blockIdx.z;
  const int n0 = blockIdx.x * 16;
  const int m0 = blockIdx.y * 128 + wave * 16;

  const int srow = tid >> 4;                 // staging row 0..15
  const int skc  = (tid & 15) * 8;           // staging k-chunk offset

  v8f acc = {0.f, 0.f, 0.f, 0.f, 0.f, 0.f, 0.f, 0.f};

  const size_t xtrow = ((size_t)b * N_ + n0 + srow) * D_;

  for (int k0 = 0; k0 < D_; k0 += 128) {
    __syncthreads();
    *(uint4*)&ldsX[srow * LDSK + skc] = *(const uint4*)&XT[xtrow + k0 + skc];
    if (k0 + 128 < D_)   // prefetch next chunk (global_prefetch_b8)
      __builtin_prefetch(&XT[xtrow + k0 + 128 + skc], 0, 1);
    __syncthreads();

#pragma unroll
    for (int ks = 0; ks < 128; ks += 32) {
      // A fragment (16-bit A 16x32): M=lane&15, K=(e>>3)*16 + 8*hi + (e&7)
      Frag16 a, xb;
      const uint4* wp =
          (const uint4*)(Wbf + (size_t)(m0 + col) * D_ + k0 + ks + 8 * hi);
      a.q[0] = wp[0];
      a.q[1] = wp[2];
      // B fragment (16-bit B 32x16): N=lane&15, K=(e>>3)*16 + 8*hi + (e&7)
      xb.q[0] = *(const uint4*)&ldsX[col * LDSK + ks + 8 * hi];
      xb.q[1] = *(const uint4*)&ldsX[col * LDSK + ks + 16 + 8 * hi];
      acc = __builtin_amdgcn_wmma_f32_16x16x32_bf16(false, a.v, false, xb.v,
                                                    (short)0, acc, false, false);
    }
  }

  // bias + store (C layout: row = r + 8*hi, col = lane&15)
#pragma unroll
  for (int r = 0; r < 8; ++r) {
    int row = m0 + r + 8 * hi;        // output channel d = hd*H + h
    int n   = n0 + col;
    float v = acc[r] + bias[row];
    if (mode == OUT_F32) {
      out_f32[((size_t)b * D_ + row) * N_ + n] = v;
    } else {
      int h = row & (H_ - 1), hd = row >> 3;
      size_t oi;
      if (mode == OUT_QK)
        oi = (((size_t)b * H_ + h) * N_ + n) * HD_ + hd;   // [b][h][n][hd]
      else
        oi = (((size_t)b * H_ + h) * HD_ + hd) * N_ + n;   // [b][h][hd][n]
      out_bf[oi] = f2bf(v);
    }
  }
}

// ---------------------------------------------------------------------------
// Fused flash-attention + head-mean of raw scores.
// Grid (N/16, B); block 256 = 8 waves; wave h handles head h.
// Mean-of-heads via conflict-free LDS slabs (b128 stores, clause-able loads).
// ---------------------------------------------------------------------------
__global__ __launch_bounds__(256) void attn_kernel(
    const unsigned short* __restrict__ Qbf,   // [B,H,N,HD] bf16
    const unsigned short* __restrict__ Kbf,   // [B,H,N,HD] bf16
    const unsigned short* __restrict__ Vbf,   // [B,H,HD,N] bf16
    unsigned short* __restrict__ XoutT,       // [B,N,D] bf16 (attention out)
    float* __restrict__ MeanOut) {            // [B,N,N]
  __shared__ float sbuf[H_][32][16];           // per-wave raw scores, 16KB
  __shared__ unsigned short pbuf[H_][16 * 32]; // per-wave P tile [n][m] bf16

  const int tid  = threadIdx.x;
  const int h    = tid >> 5, lane = tid & 31;
  const int col  = lane & 15, hi = lane >> 4;
  const int b    = blockIdx.y;
  const int n0   = blockIdx.x * 16;
  const size_t qkbase = ((size_t)b * H_ + h) * N_ * HD_;
  const size_t vbase  = ((size_t)b * H_ + h) * HD_ * N_;
  const float scale = 0.125f;                 // 1/sqrt(HD=64)

  // Q A-fragments: lane row n fixed, hd in contiguous chunks of 8.
  Frag16 aq0, aq1;
  {
    const uint4* qp = (const uint4*)(Qbf + qkbase + (size_t)(n0 + col) * HD_);
    aq0.q[0] = qp[2 * hi];      // hd = 16*hi + 0..7
    aq0.q[1] = qp[2 * hi + 1];  // hd = 16*hi + 8..15
    aq1.q[0] = qp[4 + 2 * hi];  // +32
    aq1.q[1] = qp[5 + 2 * hi];
  }

  // mean-reduction gather indices for this thread's two output positions
  const int row0 = tid >> 5,          cm0 = tid & 31;
  const int row1 = (tid + 256) >> 5,  cm1 = tid & 31;
  const int gl0 = (cm0 & 15) + 16 * (row0 >> 3), ge0 = (row0 & 7) + 8 * (cm0 >> 4);
  const int gl1 = (cm1 & 15) + 16 * (row1 >> 3), ge1 = (row1 & 7) + 8 * (cm1 >> 4);

  float rm[8], rl[8];
  v8f o[4];
  const v8f vzero = {0.f, 0.f, 0.f, 0.f, 0.f, 0.f, 0.f, 0.f};
#pragma unroll
  for (int r = 0; r < 8; ++r) { rm[r] = -1e30f; rl[r] = 0.f; }
#pragma unroll
  for (int g = 0; g < 4; ++g) o[g] = vzero;

  for (int mp = 0; mp < N_ / 32; ++mp) {
    const int mbase = mp * 32;

    // K B-fragments: lane col = m, K = hd = 16*hi + e -> contiguous 16.
    Frag16 bk00, bk01, bk10, bk11;
    {
      const uint4* kp0 =
          (const uint4*)(Kbf + qkbase + (size_t)(mbase + col) * HD_ + 16 * hi);
      const uint4* kp1 =
          (const uint4*)(Kbf + qkbase + (size_t)(mbase + 16 + col) * HD_ + 16 * hi);
      bk00.q[0] = kp0[0]; bk00.q[1] = kp0[1];   // hd 0..31 slice
      bk01.q[0] = kp0[4]; bk01.q[1] = kp0[5];   // hd 32..63
      bk10.q[0] = kp1[0]; bk10.q[1] = kp1[1];
      bk11.q[0] = kp1[4]; bk11.q[1] = kp1[5];
    }

    v8f s0 = vzero, s1 = vzero;
    s0 = __builtin_amdgcn_wmma_f32_16x16x32_bf16(false, aq0.v, false, bk00.v,
                                                 (short)0, s0, false, false);
    s0 = __builtin_amdgcn_wmma_f32_16x16x32_bf16(false, aq1.v, false, bk01.v,
                                                 (short)0, s0, false, false);
    s1 = __builtin_amdgcn_wmma_f32_16x16x32_bf16(false, aq0.v, false, bk10.v,
                                                 (short)0, s1, false, false);
    s1 = __builtin_amdgcn_wmma_f32_16x16x32_bf16(false, aq1.v, false, bk11.v,
                                                 (short)0, s1, false, false);
#pragma unroll
    for (int r = 0; r < 8; ++r) { s0[r] *= scale; s1[r] *= scale; }

    // ---- head-mean: conflict-free two-phase LDS reduction ----
    float* slab = &sbuf[h][lane][0];
    *(float4*)&slab[0]  = make_float4(s0[0], s0[1], s0[2], s0[3]);
    *(float4*)&slab[4]  = make_float4(s0[4], s0[5], s0[6], s0[7]);
    *(float4*)&slab[8]  = make_float4(s1[0], s1[1], s1[2], s1[3]);
    *(float4*)&slab[12] = make_float4(s1[4], s1[5], s1[6], s1[7]);
    __syncthreads();
    {
      float a0 = 0.f, a1 = 0.f;
#pragma unroll
      for (int w = 0; w < H_; ++w) {
        a0 += sbuf[w][gl0][ge0];
        a1 += sbuf[w][gl1][ge1];
      }
      const size_t mrow = (size_t)b * N_ * N_ + (size_t)n0 * N_ + mbase;
      MeanOut[mrow + (size_t)row0 * N_ + cm0] = a0 * 0.125f;   // 1/H
      MeanOut[mrow + (size_t)row1 * N_ + cm1] = a1 * 0.125f;
    }
    __syncthreads();   // sbuf reusable next iteration

    // ---- online softmax (rows live in 16-lane halves, matching C layout) ----
#pragma unroll
    for (int r = 0; r < 8; ++r) {
      float tmax = fmaxf(s0[r], s1[r]);
#pragma unroll
      for (int msk = 1; msk < 16; msk <<= 1)
        tmax = fmaxf(tmax, __shfl_xor(tmax, msk, 32));
      float nm   = fmaxf(rm[r], tmax);
      float corr = __expf(rm[r] - nm);
      float p0   = __expf(s0[r] - nm);
      float p1   = __expf(s1[r] - nm);
      float ts   = p0 + p1;
#pragma unroll
      for (int msk = 1; msk < 16; msk <<= 1)
        ts += __shfl_xor(ts, msk, 32);
      rl[r] = rl[r] * corr + ts;
      rm[r] = nm;
      o[0][r] *= corr; o[1][r] *= corr; o[2][r] *= corr; o[3][r] *= corr;
      int row = r + 8 * hi;
      pbuf[h][row * 32 + col]      = f2bf(p0);
      pbuf[h][row * 32 + 16 + col] = f2bf(p1);
    }

    // Re-layout P (C-layout f32 -> A-layout bf16) via this wave's LDS region.
    Frag16 ap;
    ap.q[0] = *(const uint4*)(&pbuf[h][col * 32 + 16 * hi]);
    ap.q[1] = *(const uint4*)(&pbuf[h][col * 32 + 16 * hi + 8]);

    // V B-fragments: lane hd fixed, K = m = mbase + 16*hi + e -> contiguous.
#pragma unroll
    for (int g = 0; g < 4; ++g) {
      Frag16 bv;
      int hd = g * 16 + col;
      const uint4* vp =
          (const uint4*)(Vbf + vbase + (size_t)hd * N_ + mbase + 16 * hi);
      bv.q[0] = vp[0];
      bv.q[1] = vp[1];
      o[g] = __builtin_amdgcn_wmma_f32_16x16x32_bf16(false, ap.v, false, bv.v,
                                                     (short)0, o[g], false,
                                                     false);
    }

    if (mbase + 32 < N_) {  // prefetch next K/V slabs for this head
      __builtin_prefetch(Kbf + qkbase + (size_t)(mbase + 32 + col) * HD_, 0, 1);
      __builtin_prefetch(Vbf + vbase + (size_t)(16 * hi + col) * N_ + mbase + 32, 0, 1);
    }
  }

  // normalize and write attention output in XT layout [B,N,D] (bf16)
#pragma unroll
  for (int g = 0; g < 4; ++g) {
#pragma unroll
    for (int r = 0; r < 8; ++r) {
      int n  = n0 + r + 8 * hi;
      int hd = g * 16 + col;
      float val = o[g][r] / rl[r];
      XoutT[((size_t)b * N_ + n) * D_ + (hd * H_ + h)] = f2bf(val);
    }
  }
}

// ---------------------------------------------------------------------------
// Host-side launch
// ---------------------------------------------------------------------------
extern "C" void kernel_launch(void* const* d_in, const int* in_sizes, int n_in,
                              void* d_out, int out_size, void* d_ws,
                              size_t ws_size, hipStream_t stream) {
  (void)in_sizes; (void)n_in; (void)out_size; (void)ws_size;
  const float* q_in = (const float*)d_in[0];
  const float* k_in = (const float*)d_in[1];
  const float* v_in = (const float*)d_in[2];
  const float* W    = (const float*)d_in[3];
  const float* bias = (const float*)d_in[4];

  float* merged     = (float*)d_out;                          // [B,D,N]
  float* meanscores = (float*)d_out + (size_t)B_ * D_ * N_;   // [B,N,N]

  const size_t BDN = (size_t)B_ * D_ * N_;                    // 4,194,304
  char* ws = (char*)d_ws;
  unsigned short* wbf  = (unsigned short*)(ws);
  unsigned short* inbf = (unsigned short*)(ws + 524288);                 // XT staging
  unsigned short* qbf  = (unsigned short*)(ws + 524288 + 1 * BDN * 2);
  unsigned short* kbf  = (unsigned short*)(ws + 524288 + 2 * BDN * 2);
  unsigned short* vbf  = (unsigned short*)(ws + 524288 + 3 * BDN * 2);
  unsigned short* xbf  = (unsigned short*)(ws + 524288 + 4 * BDN * 2);   // XT layout

  const dim3 gblk(256);
  const dim3 ggrid(N_ / 16, D_ / 128, B_);
  const dim3 tgrid(N_ / 32, D_ / 32, B_);

  // W -> bf16 (row-major, K contiguous for A-fragments)
  cvt_bf16_kernel<<<(D_ * D_ / 4 + 255) / 256, gblk, 0, stream>>>(W, wbf,
                                                                  D_ * D_ / 4);

  // Q = W*q + b  ([B,H,N,HD] bf16)
  cvtT_kernel<<<tgrid, gblk, 0, stream>>>(q_in, inbf);
  gemm_bias_kernel<<<ggrid, gblk, 0, stream>>>(wbf, inbf, bias, qbf, nullptr,
                                               OUT_QK);
  // K = W*k + b  ([B,H,N,HD] bf16)
  cvtT_kernel<<<tgrid, gblk, 0, stream>>>(k_in, inbf);
  gemm_bias_kernel<<<ggrid, gblk, 0, stream>>>(wbf, inbf, bias, kbf, nullptr,
                                               OUT_QK);
  // V = W*v + b  ([B,H,HD,N] bf16)
  cvtT_kernel<<<tgrid, gblk, 0, stream>>>(v_in, inbf);
  gemm_bias_kernel<<<ggrid, gblk, 0, stream>>>(wbf, inbf, bias, vbf, nullptr,
                                               OUT_V);

  // Fused attention + head-mean scores (writes XT-layout x)
  attn_kernel<<<dim3(N_ / 16, B_), gblk, 0, stream>>>(qbf, kbf, vbf, xbf,
                                                      meanscores);

  // merged = W*x + b  (f32 directly into d_out)
  gemm_bias_kernel<<<ggrid, gblk, 0, stream>>>(wbf, xbf, bias, nullptr, merged,
                                               OUT_F32);
}
